// DenseAttention_61598420959334
// MI455X (gfx1250) — matrix operand order repaired and checked by reading, not compiled, not run on hardware
//
#include <hip/hip_runtime.h>

typedef _Float16 half_t;
typedef __attribute__((ext_vector_type(16))) _Float16 v16h;
typedef __attribute__((ext_vector_type(8)))  _Float16 v8h;
typedef __attribute__((ext_vector_type(8)))  float    v8f;

#define B_ROWS 262144
#define D_DIM  128

__device__ __forceinline__ v16h cat8(v8h lo, v8h hi) {
  return __builtin_shufflevector(lo, hi, 0,1,2,3,4,5,6,7,8,9,10,11,12,13,14,15);
}

__device__ __forceinline__ v8f splat8(float v) {
  v8f r;
#pragma unroll
  for (int i = 0; i < 8; ++i) r[i] = v;
  return r;
}

__device__ __forceinline__ void cvt_store4(half_t* d, float4 w) {
  d[0] = (half_t)w.x; d[1] = (half_t)w.y; d[2] = (half_t)w.z; d[3] = (half_t)w.w;
}

// Async global->LDS 16-byte copy (ASYNCcnt path; gfx1250 VGLOBAL encoding, GV mode).
__device__ __forceinline__ void async_copy_b128(unsigned lds_off, const void* g) {
  asm volatile("global_load_async_to_lds_b128 %0, %1, off"
               : : "v"(lds_off), "v"(g) : "memory");
}
__device__ __forceinline__ void wait_async0() {
  asm volatile("s_wait_asynccnt 0" : : : "memory");
}

// A fragment (16x32 f16, MxK) from row-major [M][128] f16 LDS matrix.
// Lanes 0-15: M=lane, K = kb+{0..7,16..23}; lanes 16-31: M=lane-16, K = kb+{8..15,24..31}.
__device__ __forceinline__ v16h load_a_frag(const half_t* m_, int m0, int kb, int lane) {
  const half_t* row = m_ + (m0 + (lane & 15)) * 128 + kb + ((lane & 16) ? 8 : 0);
  v8h lo = *(const v8h*)(row);
  v8h hi = *(const v8h*)(row + 16);
  return cat8(lo, hi);
}

// B fragment (32x16 f16, KxN) where source stores B^T row-major: src[n][k].
// Lanes 0-15: N=lane, K = kb+{0..7,16..23}; lanes 16-31: N=lane-16, K = kb+{8..15,24..31}.
__device__ __forceinline__ v16h load_bt_frag(const half_t* m_, int n0, int kb, int lane) {
  const half_t* row = m_ + (n0 + (lane & 15)) * 128 + kb + ((lane & 16) ? 8 : 0);
  v8h lo = *(const v8h*)(row);
  v8h hi = *(const v8h*)(row + 16);
  return cat8(lo, hi);
}

__device__ __forceinline__ v8f wmma_f16(v16h a, v16h b, v8f c) {
  return __builtin_amdgcn_wmma_f32_16x16x32_f16(false, a, false, b, (short)0, c, false, false);
}

// ---------------- Kernel 0: zero the scores accumulator ----------------
__global__ void zero_scores_kernel(float* __restrict__ scores) {
  scores[blockIdx.x * 256 + threadIdx.x] = 0.0f;
}

// ---------------- Kernel 1: q,k,v GEMMs + partial q^T k ----------------
// 256 blocks x 256 threads (8 waves). Each block processes 8 tiles of 128 rows.
__global__ __launch_bounds__(256) void qkv_scores_kernel(
    const float* __restrict__ x,
    const float* __restrict__ Wq, const float* __restrict__ bq,
    const float* __restrict__ Wk, const float* __restrict__ bk,
    const float* __restrict__ Wv, const float* __restrict__ bv,
    half_t* __restrict__ v_out, float* __restrict__ scores) {
  __shared__ half_t xs[128 * 128];        // x tile, row-major (A source)
  __shared__ half_t wLds[3][128 * 128];   // Wq/Wk/Wv row-major [out][in] (B^T source)
  __shared__ half_t qT[128 * 128];        // q^T: [H][rowInTile]
  __shared__ half_t kT[128 * 128];        // k^T: [H][rowInTile]

  const int tid  = threadIdx.x;
  const int lane = tid & 31;
  const int wave = tid >> 5;
  const int m0 = wave * 16;
  const int nlane = lane & 15;
  const int madd  = (lane & 16) ? 8 : 0;

  const float* Ws[3] = {Wq, Wk, Wv};
#pragma unroll
  for (int m = 0; m < 3; ++m) {
    const float4* W4 = (const float4*)Ws[m];
#pragma unroll
    for (int i = 0; i < 16; ++i) {
      int idx = i * 256 + tid;
      cvt_store4(&wLds[m][idx * 4], W4[idx]);
    }
  }
  __syncthreads();

  v8f sacc[8];
#pragma unroll
  for (int nt = 0; nt < 8; ++nt) sacc[nt] = splat8(0.0f);

  for (int g = 0; g < 8; ++g) {
    const int row0 = (blockIdx.x * 8 + g) * 128;
    const float4* x4 = (const float4*)(x + (size_t)row0 * 128);
#pragma unroll
    for (int i = 0; i < 16; ++i) {
      int idx = i * 256 + tid;
      cvt_store4(&xs[idx * 4], x4[idx]);
    }
    // Warm L2 for the next x tile (global_prefetch_b8), one cacheline-sized slice per thread.
    if (g < 7) {
      const char* nxt = (const char*)(x + (size_t)(row0 + 128) * 128);
      __builtin_prefetch(nxt + tid * 256, 0, 1);
    }
    __syncthreads();

    // q (mat=0) and k (mat=1): D-tiles written transposed (packed 16B stores).
#pragma unroll
    for (int mat = 0; mat < 2; ++mat) {
      const float* bias = mat ? bk : bq;
      half_t* dstT = mat ? kT : qT;
      for (int nt = 0; nt < 8; ++nt) {
        const int n0 = nt * 16;
        v8f acc = splat8(bias[n0 + nlane]);
#pragma unroll
        for (int kk = 0; kk < 4; ++kk) {
          v16h a = load_a_frag(xs, m0, kk * 32, lane);
          v16h b = load_bt_frag(wLds[mat], n0, kk * 32, lane);
          acc = wmma_f16(a, b, acc);
        }
        v8h packed;
#pragma unroll
        for (int r = 0; r < 8; ++r) packed[r] = (half_t)acc[r];
        *(v8h*)(dstT + (n0 + nlane) * 128 + m0 + madd) = packed;
      }
    }

    // v: stream to global workspace (f16 row-major [B][128]); coalesced across lanes.
    for (int nt = 0; nt < 8; ++nt) {
      const int n0 = nt * 16;
      v8f acc = splat8(bv[n0 + nlane]);
#pragma unroll
      for (int kk = 0; kk < 4; ++kk) {
        v16h a = load_a_frag(xs, m0, kk * 32, lane);
        v16h b = load_bt_frag(wLds[2], n0, kk * 32, lane);
        acc = wmma_f16(a, b, acc);
      }
#pragma unroll
      for (int r = 0; r < 8; ++r) {
        int mrow = m0 + r + madd;
        v_out[(size_t)(row0 + mrow) * 128 + n0 + nlane] = (half_t)acc[r];
      }
    }
    __syncthreads();

    // Partial scores: S[m,n] += sum_rows q[row,m]*k[row,n] over this 128-row tile.
    for (int nt = 0; nt < 8; ++nt) {
#pragma unroll
      for (int kk = 0; kk < 4; ++kk) {
        v16h a = load_a_frag(qT, m0, kk * 32, lane);
        v16h b = load_bt_frag(kT, nt * 16, kk * 32, lane);
        sacc[nt] = wmma_f16(a, b, sacc[nt]);
      }
    }
    __syncthreads();
  }

  // Fold this block's partial S into global accumulator.
#pragma unroll
  for (int nt = 0; nt < 8; ++nt) {
    int n = nt * 16 + nlane;
#pragma unroll
    for (int r = 0; r < 8; ++r) {
      atomicAdd(&scores[(m0 + r + madd) * 128 + n], sacc[nt][r]);
    }
  }
}

// ---------------- Kernel 2: softmax over 128x128 scores -> f16 weights ----------------
__global__ __launch_bounds__(128) void softmax_kernel(const float* __restrict__ scores,
                                                      half_t* __restrict__ weights) {
  const int row = threadIdx.x;
  const float scale = 0.088388347648318447f;  // 1/sqrt(128)
  float vals[128];
  float mx = -3.4e38f;
  for (int j = 0; j < 128; ++j) {
    float v = scores[row * 128 + j] * scale;
    vals[j] = v;
    mx = fmaxf(mx, v);
  }
  float sum = 0.0f;
  for (int j = 0; j < 128; ++j) {
    float e = __expf(vals[j] - mx);
    vals[j] = e;
    sum += e;
  }
  float inv = 1.0f / sum;
  for (int j = 0; j < 128; ++j) weights[row * 128 + j] = (half_t)(vals[j] * inv);
}

// ---------------- Kernel 3: per-p: (weights @ V_p^T) @ Wo^T + bo ----------------
// 2048 blocks (p) x 256 threads. Output rows scatter to i = h*2048 + p (the flat reshape).
__global__ __launch_bounds__(256) void attn_out_kernel(
    const half_t* __restrict__ v_in, const half_t* __restrict__ weights,
    const float* __restrict__ Wo, const float* __restrict__ bo,
    float* __restrict__ out) {
  __shared__ half_t wLds[128 * 128];    // weights row-major [h][k] (A source)
  __shared__ half_t vLds[128 * 128];    // V_p row-major [j][k] (B^T source)
  __shared__ half_t woLds[128 * 128];   // Wo row-major [o][j] (B^T source)
  __shared__ half_t tLds[128 * 128];    // T row-major [h][j] (A source)

  const int tid  = threadIdx.x;
  const int lane = tid & 31;
  const int wave = tid >> 5;
  const int m0 = wave * 16;
  const int nlane = lane & 15;
  const int madd  = (lane & 16) ? 8 : 0;
  const int p = blockIdx.x;

  // Async global->LDS staging of weights (f16) and V_p (f16): pure byte copies,
  // overlapped with the Wo f32->f16 conversion below; tracked by ASYNCcnt.
  {
    const unsigned wlds0 = (unsigned)(size_t)&wLds[0];
    const unsigned vlds0 = (unsigned)(size_t)&vLds[0];
    const char* wsrc = (const char*)weights;
    const char* vsrc = (const char*)(v_in + (size_t)p * (128 * 128));
#pragma unroll
    for (int i = 0; i < 8; ++i) {
      unsigned boff = (unsigned)(i * 256 + tid) * 16u;  // 16B chunks
      async_copy_b128(wlds0 + boff, wsrc + boff);
      async_copy_b128(vlds0 + boff, vsrc + boff);
    }
  }
  const float4* Wo4 = (const float4*)Wo;
#pragma unroll
  for (int i = 0; i < 16; ++i) {
    int idx = i * 256 + tid;
    cvt_store4(&woLds[idx * 4], Wo4[idx]);
  }
  wait_async0();
  __syncthreads();

  // T = weights @ V_p^T
  for (int jt = 0; jt < 8; ++jt) {
    v8f acc = splat8(0.0f);
#pragma unroll
    for (int kk = 0; kk < 4; ++kk) {
      v16h a = load_a_frag(wLds, m0, kk * 32, lane);
      v16h b = load_bt_frag(vLds, jt * 16, kk * 32, lane);
      acc = wmma_f16(a, b, acc);
    }
    int j = jt * 16 + nlane;
#pragma unroll
    for (int r = 0; r < 8; ++r) tLds[(m0 + r + madd) * 128 + j] = (half_t)acc[r];
  }
  __syncthreads();

  // O = T @ Wo^T + bo ; out[h*2048 + p, o] = O[h, o]
  for (int ot = 0; ot < 8; ++ot) {
    int o = ot * 16 + nlane;
    v8f acc = splat8(bo[o]);
#pragma unroll
    for (int kk = 0; kk < 4; ++kk) {
      v16h a = load_a_frag(tLds, m0, kk * 32, lane);
      v16h b = load_bt_frag(woLds, ot * 16, kk * 32, lane);
      acc = wmma_f16(a, b, acc);
    }
#pragma unroll
    for (int r = 0; r < 8; ++r) {
      int h = m0 + r + madd;
      out[((size_t)h * 2048 + p) * 128 + o] = acc[r];
    }
  }
}

extern "C" void kernel_launch(void* const* d_in, const int* in_sizes, int n_in,
                              void* d_out, int out_size, void* d_ws, size_t ws_size,
                              hipStream_t stream) {
  (void)in_sizes; (void)n_in; (void)out_size; (void)ws_size;
  const float* x  = (const float*)d_in[0];
  const float* Wq = (const float*)d_in[1];
  const float* bq = (const float*)d_in[2];
  const float* Wk = (const float*)d_in[3];
  const float* bk = (const float*)d_in[4];
  const float* Wv = (const float*)d_in[5];
  const float* bv = (const float*)d_in[6];
  const float* Wo = (const float*)d_in[7];
  const float* bo = (const float*)d_in[8];
  float* out = (float*)d_out;

  // Workspace layout: [v f16: B*128] [scores f32: 128*128] [weights f16: 128*128]
  half_t* v_ws = (half_t*)d_ws;
  const size_t v_bytes = (size_t)B_ROWS * D_DIM * sizeof(half_t);
  float*  scores  = (float*)((char*)d_ws + v_bytes);
  half_t* weights = (half_t*)((char*)d_ws + v_bytes + (size_t)D_DIM * D_DIM * sizeof(float));

  zero_scores_kernel<<<64, 256, 0, stream>>>(scores);
  qkv_scores_kernel<<<256, 256, 0, stream>>>(x, Wq, bq, Wk, bk, Wv, bv, v_ws, scores);
  softmax_kernel<<<1, 128, 0, stream>>>(scores, weights);
  attn_out_kernel<<<2048, 256, 0, stream>>>(v_ws, weights, Wo, bo, out);
}
